// GeometricConstructorLegacy_1125281431924
// MI455X (gfx1250) — matrix operand order repaired and checked by reading, not compiled, not run on hardware
//
#include <hip/hip_runtime.h>
#include <hip/hip_bf16.h>

typedef __attribute__((ext_vector_type(16))) _Float16 v16h;
typedef __attribute__((ext_vector_type(8)))  _Float16 v8h;
typedef __attribute__((ext_vector_type(8)))  float    v8f;

#define BROWS 65536
#define MT 128            // rows per workgroup (8 waves x 16 rows)
#define GRID (BROWS / MT) // 512

// ---------------------------------------------------------------------------
// Pack a row-major f32 weight [fin x fout] into WMMA B-matrix lane layout:
// blocks of (32 lanes x 16 halfs), block index = kt*nnt + nt.
// lane L: n = nt*16 + (L&15); element e: k = kt*32 + (L>=16?8:0) + (e&7) + (e>=8?16:0).
// Out-of-range (k,n) -> 0 (gives exact zero padding for 132->160 etc.)
// ---------------------------------------------------------------------------
__global__ void pack_w_kernel(const float* __restrict__ W, _Float16* __restrict__ out,
                              int fin, int fout, int nkt, int nnt) {
  int idx = blockIdx.x * 256 + threadIdx.x;
  int total = nkt * nnt * 512;
  if (idx >= total) return;
  int blk = idx >> 9, ib = idx & 511;
  int L = ib >> 4, e = ib & 15;
  int kt = blk / nnt, nt = blk % nnt;
  int n = nt * 16 + (L & 15);
  int k = kt * 32 + ((L >= 16) ? 8 : 0) + (e & 7) + ((e >= 8) ? 16 : 0);
  float v = (k < fin && n < fout) ? W[(size_t)k * fout + n] : 0.0f;
  out[idx] = (_Float16)v;
}

// ssum = a + 2*b + c  (for the 4-component aggregation d_l1 + d_c1 + d_c2 + d_l2)
__global__ void sum3_kernel(const _Float16* __restrict__ a, const _Float16* __restrict__ b,
                            const _Float16* __restrict__ c, _Float16* __restrict__ o, int n) {
  int i = blockIdx.x * 256 + threadIdx.x;
  if (i >= n) return;
  o[i] = (_Float16)((float)a[i] + 2.0f * (float)b[i] + (float)c[i]);
}

// ---------------------------------------------------------------------------
// Fused GEMM: D = act( alpha * (concat(A0,A1) @ W) + beta * bias )
// M=65536, K = 32*(nkt0+nkt1), N = NT*16. 8 waves/WG, wave w handles rows
// [blk*128 + w*16, +16) and all NT n-tiles via v_wmma_f32_16x16x32_f16.
// Writes f16 activation (ld = NT*16) + up to two f32 output slots (N=128).
// ---------------------------------------------------------------------------
template <int NT, bool RELU, bool A0F32>
__global__ __launch_bounds__(256) void gemm_kernel(
    const void* __restrict__ A0v, int lda0, int nkt0,
    const _Float16* __restrict__ A1, int lda1, int nkt1,
    const _Float16* __restrict__ Wp, const float* __restrict__ bias, int fout,
    float alpha, float beta,
    _Float16* __restrict__ outH, int ldo,
    float* __restrict__ outF0, float* __restrict__ outF1) {
  const int wave = threadIdx.x >> 5;
  const int lane = threadIdx.x & 31;
  const int mloc = lane & 15;
  const int koff = (lane >> 4) << 3;          // 0 or 8 (A layout half)
  const size_t rowA = (size_t)blockIdx.x * MT + wave * 16 + mloc;

  v8f acc[NT];
#pragma unroll
  for (int nt = 0; nt < NT; ++nt) acc[nt] = (v8f){0.f,0.f,0.f,0.f,0.f,0.f,0.f,0.f};

  const _Float16* A0h = (const _Float16*)A0v;
  const float*    A0f = (const float*)A0v;

  for (int kt = 0; kt < nkt0; ++kt) {
    v16h a;
    if (A0F32) {
      const float* pa = A0f + rowA * lda0 + kt * 32 + koff;
#pragma unroll
      for (int i = 0; i < 8; ++i) { a[i] = (_Float16)pa[i]; a[i + 8] = (_Float16)pa[i + 16]; }
    } else {
      const _Float16* pa = A0h + rowA * lda0 + kt * 32 + koff;
      v8h lo = *reinterpret_cast<const v8h*>(pa);
      v8h hi = *reinterpret_cast<const v8h*>(pa + 16);
      a = __builtin_shufflevector(lo, hi, 0,1,2,3,4,5,6,7,8,9,10,11,12,13,14,15);
    }
#pragma unroll
    for (int nt = 0; nt < NT; ++nt) {
      const _Float16* pb = Wp + (((size_t)kt * NT + nt) << 9) + (lane << 4);
      v8h blo = *reinterpret_cast<const v8h*>(pb);
      v8h bhi = *reinterpret_cast<const v8h*>(pb + 8);
      v16h b = __builtin_shufflevector(blo, bhi, 0,1,2,3,4,5,6,7,8,9,10,11,12,13,14,15);
      acc[nt] = __builtin_amdgcn_wmma_f32_16x16x32_f16(false, a, false, b, (short)0,
                                                       acc[nt], false, false);
    }
  }
  for (int j = 0; j < nkt1; ++j) {
    const int kt = nkt0 + j;
    const _Float16* pa = A1 + rowA * lda1 + j * 32 + koff;
    v8h lo = *reinterpret_cast<const v8h*>(pa);
    v8h hi = *reinterpret_cast<const v8h*>(pa + 16);
    v16h a = __builtin_shufflevector(lo, hi, 0,1,2,3,4,5,6,7,8,9,10,11,12,13,14,15);
#pragma unroll
    for (int nt = 0; nt < NT; ++nt) {
      const _Float16* pb = Wp + (((size_t)kt * NT + nt) << 9) + (lane << 4);
      v8h blo = *reinterpret_cast<const v8h*>(pb);
      v8h bhi = *reinterpret_cast<const v8h*>(pb + 8);
      v16h b = __builtin_shufflevector(blo, bhi, 0,1,2,3,4,5,6,7,8,9,10,11,12,13,14,15);
      acc[nt] = __builtin_amdgcn_wmma_f32_16x16x32_f16(false, a, false, b, (short)0,
                                                       acc[nt], false, false);
    }
  }

  // Epilogue: D layout — lane holds N = lane&15; VGPR r holds M = r + (lane>=16)*8
  const size_t rbase = (size_t)blockIdx.x * MT + wave * 16 + ((lane >> 4) << 3);
#pragma unroll
  for (int nt = 0; nt < NT; ++nt) {
    const int n = nt * 16 + mloc;
    const float bv = (n < fout) ? beta * bias[n] : 0.0f;
#pragma unroll
    for (int r = 0; r < 8; ++r) {
      float v = alpha * acc[nt][r] + bv;
      if (RELU) v = v > 0.0f ? v : 0.0f;
      const size_t mrow = rbase + r;
      outH[mrow * (size_t)ldo + n] = (_Float16)v;
      if (outF0 && n < 128) outF0[mrow * 128 + n] = v;
      if (outF1 && n < 128) outF1[mrow * 128 + n] = v;
    }
  }
}

// ---------------------------------------------------------------------------
// Host-side helpers
// ---------------------------------------------------------------------------
static void launch_gemm(hipStream_t st, int NT, bool relu, bool a0f32,
                        const void* A0, int lda0, int nkt0,
                        const _Float16* A1, int lda1, int nkt1,
                        const _Float16* Wp, const float* bias, int fout,
                        float alpha, float beta,
                        _Float16* outH, int ldo, float* oF0, float* oF1) {
  dim3 g(GRID), b(256);
  if (NT == 10 && relu && a0f32)
    gemm_kernel<10, true, true><<<g, b, 0, st>>>(A0, lda0, nkt0, A1, lda1, nkt1, Wp, bias,
                                                 fout, alpha, beta, outH, ldo, oF0, oF1);
  else if (NT == 10 && relu)
    gemm_kernel<10, true, false><<<g, b, 0, st>>>(A0, lda0, nkt0, A1, lda1, nkt1, Wp, bias,
                                                  fout, alpha, beta, outH, ldo, oF0, oF1);
  else
    gemm_kernel<8, false, false><<<g, b, 0, st>>>(A0, lda0, nkt0, A1, lda1, nkt1, Wp, bias,
                                                  fout, alpha, beta, outH, ldo, oF0, oF1);
}

static void launch_pack(hipStream_t st, const float* W, _Float16* out,
                        int fin, int fout, int nkt, int nnt) {
  int total = nkt * nnt * 512;
  pack_w_kernel<<<(total + 255) / 256, 256, 0, st>>>(W, out, fin, fout, nkt, nnt);
}

extern "C" void kernel_launch(void* const* d_in, const int* in_sizes, int n_in,
                              void* d_out, int out_size, void* d_ws, size_t ws_size,
                              hipStream_t stream) {
  (void)in_sizes; (void)n_in; (void)out_size; (void)ws_size;
  const float* signal   = (const float*)d_in[0];
  const float* pp_msg_w = (const float*)d_in[1];  const float* pp_msg_b = (const float*)d_in[2];
  const float* pp_upd_w = (const float*)d_in[3];  const float* pp_upd_b = (const float*)d_in[4];
  const float* ppf_w0 = (const float*)d_in[5];  const float* ppf_b0 = (const float*)d_in[6];
  const float* ppf_w1 = (const float*)d_in[7];  const float* ppf_b1 = (const float*)d_in[8];
  const float* ppf_w2 = (const float*)d_in[9];  const float* ppf_b2 = (const float*)d_in[10];
  const float* ln_w0 = (const float*)d_in[11]; const float* ln_b0 = (const float*)d_in[12];
  const float* ln_w1 = (const float*)d_in[13]; const float* ln_b1 = (const float*)d_in[14];
  const float* ln_w2 = (const float*)d_in[15]; const float* ln_b2 = (const float*)d_in[16];
  const float* ln_w3 = (const float*)d_in[17]; const float* ln_b3 = (const float*)d_in[18];
  const float* cr_w0 = (const float*)d_in[19]; const float* cr_b0 = (const float*)d_in[20];
  const float* cr_w1 = (const float*)d_in[21]; const float* cr_b1 = (const float*)d_in[22];
  const float* cr_w2 = (const float*)d_in[23]; const float* cr_b2 = (const float*)d_in[24];
  const float* cr_w3 = (const float*)d_in[25]; const float* cr_b3 = (const float*)d_in[26];
  const float* mp_msg_w = (const float*)d_in[27]; const float* mp_msg_b = (const float*)d_in[28];
  const float* mp_upd_w = (const float*)d_in[29]; const float* mp_upd_b = (const float*)d_in[30];
  const float* mpf_w0 = (const float*)d_in[31]; const float* mpf_b0 = (const float*)d_in[32];
  const float* mpf_w1 = (const float*)d_in[33]; const float* mpf_b1 = (const float*)d_in[34];
  const float* mpf_w2 = (const float*)d_in[35]; const float* mpf_b2 = (const float*)d_in[36];

  // ---- workspace carve (bump allocator, 256B aligned) ----
  char* base = (char*)d_ws; size_t off = 0;
  auto alloc = [&](size_t bytes) -> void* {
    void* p = base + off; off = (off + bytes + 255) & ~(size_t)255; return p;
  };
  auto allocH = [&](size_t elems) -> _Float16* { return (_Float16*)alloc(elems * 2); };
  const size_t NB = (size_t)BROWS;

  // packed weights
  _Float16* wp_ppf0 = allocH(20 * 10 * 512);
  _Float16* wp_ppf1 = allocH(5 * 10 * 512);
  _Float16* wp_ppf2 = allocH(5 * 8 * 512);
  _Float16* wp_ln0  = allocH(8 * 10 * 512);
  _Float16* wp_ln1  = allocH(5 * 10 * 512);
  _Float16* wp_ln2  = allocH(5 * 10 * 512);
  _Float16* wp_ln3  = allocH(5 * 8 * 512);
  _Float16* wp_cr0  = allocH(8 * 10 * 512);
  _Float16* wp_cr1  = allocH(5 * 10 * 512);
  _Float16* wp_cr2  = allocH(5 * 10 * 512);
  _Float16* wp_cr3  = allocH(5 * 8 * 512);
  _Float16* wp_mpf0 = allocH(8 * 10 * 512);
  _Float16* wp_mpf1 = allocH(5 * 10 * 512);
  _Float16* wp_mpf2 = allocH(5 * 8 * 512);
  _Float16* wp_ppm  = allocH(4 * 8 * 512);
  _Float16* wp_ppu  = allocH(4 * 8 * 512);
  _Float16* wp_mpm  = allocH(4 * 8 * 512);
  _Float16* wp_mpu  = allocH(4 * 8 * 512);

  // f16 activation buffers
  _Float16* h160a = allocH(NB * 160);
  _Float16* h160b = allocH(NB * 160);
  _Float16* p1  = allocH(NB * 128);
  _Float16* l1u = allocH(NB * 128);
  _Float16* c1u = allocH(NB * 128);
  _Float16* p3u = allocH(NB * 128);
  _Float16* l2u = allocH(NB * 128);
  _Float16* s128 = allocH(NB * 128);
  _Float16* r128 = allocH(NB * 128);
  _Float16* dl1 = allocH(NB * 128);
  _Float16* dl2 = allocH(NB * 128);
  _Float16* dp3 = allocH(NB * 128);
  _Float16* dc1 = allocH(NB * 128);

  float* OUT = (float*)d_out;
  auto slot = [&](int s) -> float* { return OUT + (size_t)s * NB * 128; };

  // ---- pack weights (once per launch; tiny) ----
  launch_pack(stream, ppf_w0, wp_ppf0, 640, 132, 20, 10);
  launch_pack(stream, ppf_w1, wp_ppf1, 132, 132, 5, 10);
  launch_pack(stream, ppf_w2, wp_ppf2, 132, 128, 5, 8);
  launch_pack(stream, ln_w0, wp_ln0, 256, 132, 8, 10);
  launch_pack(stream, ln_w1, wp_ln1, 132, 132, 5, 10);
  launch_pack(stream, ln_w2, wp_ln2, 132, 132, 5, 10);
  launch_pack(stream, ln_w3, wp_ln3, 132, 128, 5, 8);
  launch_pack(stream, cr_w0, wp_cr0, 256, 132, 8, 10);
  launch_pack(stream, cr_w1, wp_cr1, 132, 132, 5, 10);
  launch_pack(stream, cr_w2, wp_cr2, 132, 132, 5, 10);
  launch_pack(stream, cr_w3, wp_cr3, 132, 128, 5, 8);
  launch_pack(stream, mpf_w0, wp_mpf0, 256, 132, 8, 10);
  launch_pack(stream, mpf_w1, wp_mpf1, 132, 132, 5, 10);
  launch_pack(stream, mpf_w2, wp_mpf2, 132, 128, 5, 8);
  launch_pack(stream, pp_msg_w, wp_ppm, 128, 128, 4, 8);
  launch_pack(stream, pp_upd_w, wp_ppu, 128, 128, 4, 8);
  launch_pack(stream, mp_msg_w, wp_mpm, 128, 128, 4, 8);
  launch_pack(stream, mp_upd_w, wp_mpu, 128, 128, 4, 8);

  // ---- upward pass (up_p1==up_p2, up_c1==up_c2, up_l2==up_l3) ----
  // p1 = pp_fc(concat(signal, 0))
  launch_gemm(stream, 10, true, true,  signal, 512, 16, nullptr, 0, 0, wp_ppf0, ppf_b0, 132, 1, 1, h160a, 160, nullptr, nullptr);
  launch_gemm(stream, 10, true, false, h160a, 160, 5, nullptr, 0, 0, wp_ppf1, ppf_b1, 132, 1, 1, h160b, 160, nullptr, nullptr);
  launch_gemm(stream, 8, false, false, h160b, 160, 5, nullptr, 0, 0, wp_ppf2, ppf_b2, 128, 1, 1, p1, 128, nullptr, nullptr);
  // up_l1 = ln(concat(p1,p1))
  launch_gemm(stream, 10, true, false, p1, 128, 4, p1, 128, 4, wp_ln0, ln_b0, 132, 1, 1, h160a, 160, nullptr, nullptr);
  launch_gemm(stream, 10, true, false, h160a, 160, 5, nullptr, 0, 0, wp_ln1, ln_b1, 132, 1, 1, h160b, 160, nullptr, nullptr);
  launch_gemm(stream, 10, true, false, h160b, 160, 5, nullptr, 0, 0, wp_ln2, ln_b2, 132, 1, 1, h160a, 160, nullptr, nullptr);
  launch_gemm(stream, 8, false, false, h160a, 160, 5, nullptr, 0, 0, wp_ln3, ln_b3, 128, 1, 1, l1u, 128, nullptr, nullptr);
  // up_c1 = cr(concat(p1,p1))
  launch_gemm(stream, 10, true, false, p1, 128, 4, p1, 128, 4, wp_cr0, cr_b0, 132, 1, 1, h160a, 160, nullptr, nullptr);
  launch_gemm(stream, 10, true, false, h160a, 160, 5, nullptr, 0, 0, wp_cr1, cr_b1, 132, 1, 1, h160b, 160, nullptr, nullptr);
  launch_gemm(stream, 10, true, false, h160b, 160, 5, nullptr, 0, 0, wp_cr2, cr_b2, 132, 1, 1, h160a, 160, nullptr, nullptr);
  launch_gemm(stream, 8, false, false, h160a, 160, 5, nullptr, 0, 0, wp_cr3, cr_b3, 128, 1, 1, c1u, 128, nullptr, nullptr);
  // up_p3 = point_prop(signal, [c1,c1]):  s = 2*c1@mw + 2*mb ; r = s@uw + ub
  launch_gemm(stream, 8, false, false, c1u, 128, 4, nullptr, 0, 0, wp_ppm, pp_msg_b, 128, 2, 2, s128, 128, nullptr, nullptr);
  launch_gemm(stream, 8, false, false, s128, 128, 4, nullptr, 0, 0, wp_ppu, pp_upd_b, 128, 1, 1, r128, 128, nullptr, nullptr);
  launch_gemm(stream, 10, true, true,  signal, 512, 16, r128, 128, 4, wp_ppf0, ppf_b0, 132, 1, 1, h160a, 160, nullptr, nullptr);
  launch_gemm(stream, 10, true, false, h160a, 160, 5, nullptr, 0, 0, wp_ppf1, ppf_b1, 132, 1, 1, h160b, 160, nullptr, nullptr);
  launch_gemm(stream, 8, false, false, h160b, 160, 5, nullptr, 0, 0, wp_ppf2, ppf_b2, 128, 1, 1, p3u, 128, nullptr, nullptr);
  // up_l2 = ln(concat(p1,p3))
  launch_gemm(stream, 10, true, false, p1, 128, 4, p3u, 128, 4, wp_ln0, ln_b0, 132, 1, 1, h160a, 160, nullptr, nullptr);
  launch_gemm(stream, 10, true, false, h160a, 160, 5, nullptr, 0, 0, wp_ln1, ln_b1, 132, 1, 1, h160b, 160, nullptr, nullptr);
  launch_gemm(stream, 10, true, false, h160b, 160, 5, nullptr, 0, 0, wp_ln2, ln_b2, 132, 1, 1, h160a, 160, nullptr, nullptr);
  launch_gemm(stream, 8, false, false, h160a, 160, 5, nullptr, 0, 0, wp_ln3, ln_b3, 128, 1, 1, l2u, 128, nullptr, nullptr);

  // ---- downward pass ----
  // d_l1 = mp_fc(concat(up_l1, 0))  -> slot 2
  launch_gemm(stream, 10, true, false, l1u, 128, 4, nullptr, 0, 0, wp_mpf0, mpf_b0, 132, 1, 1, h160a, 160, nullptr, nullptr);
  launch_gemm(stream, 10, true, false, h160a, 160, 5, nullptr, 0, 0, wp_mpf1, mpf_b1, 132, 1, 1, h160b, 160, nullptr, nullptr);
  launch_gemm(stream, 8, false, false, h160b, 160, 5, nullptr, 0, 0, wp_mpf2, mpf_b2, 128, 1, 1, dl1, 128, slot(2), nullptr);
  // d_l2 = mp_fc(concat(up_l2, 0))  -> slots 6,7
  launch_gemm(stream, 10, true, false, l2u, 128, 4, nullptr, 0, 0, wp_mpf0, mpf_b0, 132, 1, 1, h160a, 160, nullptr, nullptr);
  launch_gemm(stream, 10, true, false, h160a, 160, 5, nullptr, 0, 0, wp_mpf1, mpf_b1, 132, 1, 1, h160b, 160, nullptr, nullptr);
  launch_gemm(stream, 8, false, false, h160b, 160, 5, nullptr, 0, 0, wp_mpf2, mpf_b2, 128, 1, 1, dl2, 128, slot(6), slot(7));
  // d_p3 = msg_prop(up_p3, [d_l2,d_l2]) -> slot 5
  launch_gemm(stream, 8, false, false, dl2, 128, 4, nullptr, 0, 0, wp_mpm, mp_msg_b, 128, 2, 2, s128, 128, nullptr, nullptr);
  launch_gemm(stream, 8, false, false, s128, 128, 4, nullptr, 0, 0, wp_mpu, mp_upd_b, 128, 1, 1, r128, 128, nullptr, nullptr);
  launch_gemm(stream, 10, true, false, p3u, 128, 4, r128, 128, 4, wp_mpf0, mpf_b0, 132, 1, 1, h160a, 160, nullptr, nullptr);
  launch_gemm(stream, 10, true, false, h160a, 160, 5, nullptr, 0, 0, wp_mpf1, mpf_b1, 132, 1, 1, h160b, 160, nullptr, nullptr);
  launch_gemm(stream, 8, false, false, h160b, 160, 5, nullptr, 0, 0, wp_mpf2, mpf_b2, 128, 1, 1, dp3, 128, slot(5), nullptr);
  // d_c1 = msg_prop(up_c1, [d_p3]) -> slots 3,4
  launch_gemm(stream, 8, false, false, dp3, 128, 4, nullptr, 0, 0, wp_mpm, mp_msg_b, 128, 1, 1, s128, 128, nullptr, nullptr);
  launch_gemm(stream, 8, false, false, s128, 128, 4, nullptr, 0, 0, wp_mpu, mp_upd_b, 128, 1, 1, r128, 128, nullptr, nullptr);
  launch_gemm(stream, 10, true, false, c1u, 128, 4, r128, 128, 4, wp_mpf0, mpf_b0, 132, 1, 1, h160a, 160, nullptr, nullptr);
  launch_gemm(stream, 10, true, false, h160a, 160, 5, nullptr, 0, 0, wp_mpf1, mpf_b1, 132, 1, 1, h160b, 160, nullptr, nullptr);
  launch_gemm(stream, 8, false, false, h160b, 160, 5, nullptr, 0, 0, wp_mpf2, mpf_b2, 128, 1, 1, dc1, 128, slot(3), slot(4));
  // d_p1 = msg_prop(up_p1, [d_l1, d_c1, d_c1, d_l2]) -> slots 0,1
  {
    int n = BROWS * 128;
    sum3_kernel<<<(n + 255) / 256, 256, 0, stream>>>(dl1, dc1, dl2, r128, n); // r128 = dl1 + 2*dc1 + dl2
  }
  launch_gemm(stream, 8, false, false, r128, 128, 4, nullptr, 0, 0, wp_mpm, mp_msg_b, 128, 1, 4, s128, 128, nullptr, nullptr);
  launch_gemm(stream, 8, false, false, s128, 128, 4, nullptr, 0, 0, wp_mpu, mp_upd_b, 128, 1, 1, dl2, 128, nullptr, nullptr); // reuse dl2 as r
  launch_gemm(stream, 10, true, false, p1, 128, 4, dl2, 128, 4, wp_mpf0, mpf_b0, 132, 1, 1, h160a, 160, nullptr, nullptr);
  launch_gemm(stream, 10, true, false, h160a, 160, 5, nullptr, 0, 0, wp_mpf1, mpf_b1, 132, 1, 1, h160b, 160, nullptr, nullptr);
  launch_gemm(stream, 8, false, false, h160b, 160, 5, nullptr, 0, 0, wp_mpf2, mpf_b2, 128, 1, 1, dl1, 128, slot(0), slot(1));
}
